// RPNGenerator_12816182411989
// MI455X (gfx1250) — matrix operand order repaired and checked by reading, not compiled, not run on hardware
//
#include <hip/hip_runtime.h>
#include <hip/hip_bf16.h>
#include <math.h>
#include <stdint.h>

#define BDIM 4
#define CDIM 256
#define HDIM 200
#define WDIM 304
#define ADIM 3
#define HW   (HDIM*WDIM)        /* 60800  */
#define HWA  (HW*ADIM)          /* 182400 */
#define IMG_Hf 800.0f
#define IMG_Wf 1216.0f
#define PRE_NMS 1000
#define NMS_THRESH 0.7f
#define MIN_SIZE 0.001f
#define BBOX_CLIP 4.135166556742356f   /* log(1000/16) */

typedef __attribute__((ext_vector_type(16))) __bf16 v16bf;
typedef __attribute__((ext_vector_type(8)))  float  v8f;
typedef __attribute__((ext_vector_type(4)))  unsigned int u32x4;

__device__ __forceinline__ v16bf frag_from2(const __bf16* p0, const __bf16* p1){
  union { u32x4 u[2]; v16bf v; } t;
  t.u[0] = *(const u32x4*)p0;
  t.u[1] = *(const u32x4*)p1;
  return t.v;
}

__device__ __forceinline__ unsigned lds_addr(const void* p){
  return (unsigned)(uintptr_t)p;   // flat LDS aperture: low 32 bits = DS byte address
}

// ---------------- prep: NCHW f32 -> NHWC bf16 (tiled transpose) ----------------
__global__ void k_feat_nhwc(const float* __restrict__ feat, __bf16* __restrict__ out){
  __shared__ float tile[32][33];
  const int b = blockIdx.z, c0 = blockIdx.y*32, s0 = blockIdx.x*32;
  const int tx = threadIdx.x, ty = threadIdx.y;
  #pragma unroll
  for (int i=0;i<4;++i){
    int c = c0 + ty + i*8;
    tile[ty+i*8][tx] = feat[((size_t)(b*CDIM + c))*HW + s0 + tx];
  }
  __syncthreads();
  #pragma unroll
  for (int i=0;i<4;++i){
    int s = s0 + ty + i*8;
    out[((size_t)b*HW + s)*CDIM + c0 + tx] = (__bf16)tile[tx][ty+i*8];
  }
}

// conv_w [co][ci][3][3] f32 -> wB [tap][co][ci] bf16
__global__ void k_prep_wconv(const float* __restrict__ w, __bf16* __restrict__ out){
  int o = blockIdx.x*256 + threadIdx.x;
  if (o >= 9*CDIM*CDIM) return;
  int ci = o & 255; int r = o >> 8; int co = r & 255; int tap = r >> 8;
  out[o] = (__bf16) w[(size_t)(co*CDIM + ci)*9 + tap];
}

// head weights packed as [n][ci] bf16, n: 0..2 cls, 3..14 bbox(a*4+d), 15 pad
__global__ void k_prep_whead(const float* __restrict__ cw, const float* __restrict__ bw,
                             __bf16* __restrict__ out){
  int o = blockIdx.x*256 + threadIdx.x;
  if (o >= 16*CDIM) return;
  int ci = o & 255; int n = o >> 8;
  float v = 0.f;
  if (n < 3) v = cw[n*CDIM + ci];
  else if (n < 15) v = bw[(n-3)*CDIM + ci];
  out[o] = (__bf16)v;
}

// ---------------- fused 3x3 conv + ReLU + 1x1 heads (implicit GEMM, WMMA) ------
// Staging uses CDNA5 async global->LDS copies (ASYNCcnt) overlapped with WMMA.
__global__ void __launch_bounds__(256, 1) k_conv_fused(
    const __bf16* __restrict__ feat, const __bf16* __restrict__ wB,
    const __bf16* __restrict__ wH, const float* __restrict__ convB,
    const float* __restrict__ clsB, const float* __restrict__ boxB,
    float* __restrict__ scores, float* __restrict__ deltas)
{
  __shared__ __align__(16) __bf16 Abuf[2][64*32];
  __shared__ __align__(16) __bf16 Bbuf[2][256*32];
  __shared__ __align__(16) __bf16 Tbuf[64*256];

  const int tid  = threadIdx.x;
  const int lane = tid & 31, wid = tid >> 5;
  const int lq = lane & 15, half = lane >> 4;
  const int mb = wid & 3, nh = wid >> 2;        // wave tile: 16M x 128N

  const int base = blockIdx.x * 64;             // 64 positions per workgroup
  const int b  = base / HW;
  const int r0 = base - b*HW;

  // A-staging coordinates (thread moves 8 bf16 = 16B per stage)
  const int am = tid >> 2, aq = tid & 3;
  const int ar = r0 + am;
  const int ah = ar / WDIM, aw = ar - ah*WDIM;

  const unsigned long long featAddr = (unsigned long long)(uintptr_t)feat;
  const unsigned long long wbAddr   = (unsigned long long)(uintptr_t)wB;

  const u32x4 vzero = {};
  const v8f fzero = {};
  v8f acc[8];
  #pragma unroll
  for (int i=0;i<8;++i) acc[i] = fzero;

  auto issueStage = [&](int s, int bi){
    int tap = s >> 3, kk = s & 7;
    int dy = tap/3 - 1, dx = tap - (tap/3)*3 - 1;
    int hh = ah + dy, ww = aw + dx;
    bool ok = ((unsigned)hh < (unsigned)HDIM) && ((unsigned)ww < (unsigned)WDIM);
    unsigned ldsA = lds_addr(&Abuf[bi][am*32 + aq*8]);
    if (ok){
      unsigned goff = (unsigned)(((((unsigned)b*HW + (unsigned)hh*WDIM + ww) << 8)
                                  + kk*32 + aq*8) * 2u);
      asm volatile("global_load_async_to_lds_b128 %0, %1, %2 offset:0"
                   :: "v"(ldsA), "v"(goff), "s"(featAddr) : "memory");
    } else {
      *(u32x4*)(&Abuf[bi][am*32 + aq*8]) = vzero;     // zero-pad halo
    }
    unsigned ldsB = lds_addr(&Bbuf[bi][tid*32]);
    unsigned boff = (unsigned)(((((unsigned)tap*256 + tid) << 8) + kk*32) * 2u);
    asm volatile(
      "global_load_async_to_lds_b128 %0, %1, %2 offset:0\n"
      "global_load_async_to_lds_b128 %0, %1, %2 offset:16\n"
      "global_load_async_to_lds_b128 %0, %1, %2 offset:32\n"
      "global_load_async_to_lds_b128 %0, %1, %2 offset:48"
      :: "v"(ldsB), "v"(boff), "s"(wbAddr) : "memory");
  };

  issueStage(0, 0);
  asm volatile("s_wait_asynccnt 0x0" ::: "memory");
  __syncthreads();

  for (int s=0; s<72; ++s){                    // 9 taps x 8 channel chunks, K=32 each
    if (s+1 < 72) issueStage(s+1, (s+1)&1);    // async copy next stage while computing
    const __bf16* Ab = &Abuf[s&1][0];
    const __bf16* Bb = &Bbuf[s&1][0];
    const int mrow = mb*16 + lq;
    const int nbase = nh*128 + lq;
    v16bf af = frag_from2(Ab + mrow*32 + half*16, Ab + mrow*32 + half*16 + 8);
    // load all 8 B-fragments into distinct registers, then issue WMMAs back-to-back
    v16bf b0 = frag_from2(Bb + (nbase+  0)*32 + half*16, Bb + (nbase+  0)*32 + half*16 + 8);
    v16bf b1 = frag_from2(Bb + (nbase+ 16)*32 + half*16, Bb + (nbase+ 16)*32 + half*16 + 8);
    v16bf b2 = frag_from2(Bb + (nbase+ 32)*32 + half*16, Bb + (nbase+ 32)*32 + half*16 + 8);
    v16bf b3 = frag_from2(Bb + (nbase+ 48)*32 + half*16, Bb + (nbase+ 48)*32 + half*16 + 8);
    v16bf b4 = frag_from2(Bb + (nbase+ 64)*32 + half*16, Bb + (nbase+ 64)*32 + half*16 + 8);
    v16bf b5 = frag_from2(Bb + (nbase+ 80)*32 + half*16, Bb + (nbase+ 80)*32 + half*16 + 8);
    v16bf b6 = frag_from2(Bb + (nbase+ 96)*32 + half*16, Bb + (nbase+ 96)*32 + half*16 + 8);
    v16bf b7 = frag_from2(Bb + (nbase+112)*32 + half*16, Bb + (nbase+112)*32 + half*16 + 8);
    acc[0] = __builtin_amdgcn_wmma_f32_16x16x32_bf16(false, af, false, b0, (short)0, acc[0], false, false);
    acc[1] = __builtin_amdgcn_wmma_f32_16x16x32_bf16(false, af, false, b1, (short)0, acc[1], false, false);
    acc[2] = __builtin_amdgcn_wmma_f32_16x16x32_bf16(false, af, false, b2, (short)0, acc[2], false, false);
    acc[3] = __builtin_amdgcn_wmma_f32_16x16x32_bf16(false, af, false, b3, (short)0, acc[3], false, false);
    acc[4] = __builtin_amdgcn_wmma_f32_16x16x32_bf16(false, af, false, b4, (short)0, acc[4], false, false);
    acc[5] = __builtin_amdgcn_wmma_f32_16x16x32_bf16(false, af, false, b5, (short)0, acc[5], false, false);
    acc[6] = __builtin_amdgcn_wmma_f32_16x16x32_bf16(false, af, false, b6, (short)0, acc[6], false, false);
    acc[7] = __builtin_amdgcn_wmma_f32_16x16x32_bf16(false, af, false, b7, (short)0, acc[7], false, false);
    asm volatile("s_wait_asynccnt 0x0" ::: "memory");
    __syncthreads();
  }

  // bias + ReLU, park tile in LDS as bf16 (t never touches global memory)
  #pragma unroll
  for (int nb=0; nb<8; ++nb){
    int n = nh*128 + nb*16 + lq;
    float cb = convB[n];
    #pragma unroll
    for (int i=0;i<8;++i){
      float v = acc[nb][i] + cb;
      v = v > 0.f ? v : 0.f;
      int m = mb*16 + i + half*8;
      Tbuf[m*256 + n] = (__bf16)v;
    }
  }
  __syncthreads();

  // head GEMM: (64 x 256) x (256 x 16) -> objectness + deltas
  if (wid < 4){
    v8f hacc = fzero;
    #pragma unroll
    for (int kk=0; kk<8; ++kk){
      const __bf16* tb = &Tbuf[(mb*16 + lq)*256 + kk*32];
      v16bf af = frag_from2(tb + half*16, tb + half*16 + 8);
      const __bf16* hb = wH + lq*256 + kk*32 + half*16;
      v16bf bf = frag_from2(hb, hb + 8);
      hacc = __builtin_amdgcn_wmma_f32_16x16x32_bf16(
          false, af, false, bf, (short)0, hacc, false, false);
    }
    int n = lq;
    #pragma unroll
    for (int i=0;i<8;++i){
      int m = mb*16 + i + half*8;
      int hw = r0 + m;
      float v = hacc[i];
      if (n < 3){
        scores[(size_t)b*HWA + (size_t)hw*3 + n] = v + clsB[n];
      } else if (n < 15){
        int q = n - 3;
        deltas[(((size_t)b*HWA + (size_t)hw*3 + (q>>2)) << 2) + (q & 3)] = v + boxB[q];
      }
    }
  }
}

// ---------------- top-k (radix select + bitonic sort), one block per batch ----
__device__ __forceinline__ unsigned monoKey(float f){
  unsigned u = __float_as_uint(f);
  return (u & 0x80000000u) ? ~u : (u | 0x80000000u);
}
__device__ __forceinline__ float imonoKey(unsigned k){
  unsigned u = (k & 0x80000000u) ? (k & 0x7FFFFFFFu) : ~k;
  return __uint_as_float(u);
}

__global__ void k_topk(const float* __restrict__ scores, int* __restrict__ topIdx,
                       float* __restrict__ topLogit){
  const int b = blockIdx.x, tid = threadIdx.x;
  const float* s = scores + (size_t)b*HWA;
  __shared__ unsigned hist[256];
  __shared__ unsigned sPrefix, sKneed;
  __shared__ int sCnt;
  __shared__ unsigned long long cand[2048];
  if (tid==0){ sPrefix=0u; sKneed=PRE_NMS; sCnt=0; }
  __syncthreads();
  for (int pass=0; pass<3; ++pass){
    int shift = 24 - 8*pass;
    hist[tid] = 0u;
    __syncthreads();
    unsigned pref = sPrefix;
    for (int i=tid; i<HWA; i+=256){
      unsigned k = monoKey(s[i]);
      bool match = (pass==0) || ((k >> (shift+8)) == (pref >> (shift+8)));
      if (match) atomicAdd(&hist[(k>>shift)&255u], 1u);
    }
    __syncthreads();
    if (tid==0){
      unsigned cum=0, need=sKneed; int sel=0;
      for (int v=255; v>=0; --v){
        unsigned c = hist[v];
        if (cum + c >= need){ sel = v; need -= cum; break; }
        cum += c;
      }
      sKneed = need;
      sPrefix = pref | ((unsigned)sel << shift);
    }
    __syncthreads();
  }
  const unsigned thr = sPrefix >> 8;
  for (int i=tid; i<HWA; i+=256){
    unsigned k = monoKey(s[i]);
    if ((k>>8) >= thr){
      int pos = atomicAdd(&sCnt, 1);
      if (pos < 2048)
        cand[pos] = ((unsigned long long)k << 32) | (unsigned long long)(0xFFFFFFFFu - (unsigned)i);
    }
  }
  __syncthreads();
  int cnt = sCnt; if (cnt > 2048) cnt = 2048;
  for (int i=tid; i<2048; i+=256) if (i >= cnt) cand[i] = 0ull;
  // bitonic sort, descending (key desc, index asc via complemented low word)
  for (int kk=2; kk<=2048; kk<<=1){
    for (int j=kk>>1; j>0; j>>=1){
      __syncthreads();
      for (int t=tid; t<2048; t+=256){
        int ixj = t ^ j;
        if (ixj > t){
          unsigned long long a = cand[t], c = cand[ixj];
          bool ascBlock = ((t & kk) != 0);
          bool doSwap = ascBlock ? (a > c) : (a < c);
          if (doSwap){ cand[t]=c; cand[ixj]=a; }
        }
      }
    }
  }
  __syncthreads();
  for (int i=tid; i<PRE_NMS; i+=256){
    unsigned long long c = cand[i];
    unsigned k = (unsigned)(c >> 32);
    int idx = (int)(0xFFFFFFFFu - (unsigned)(c & 0xFFFFFFFFull));
    topIdx[b*PRE_NMS + i]   = idx;
    topLogit[b*PRE_NMS + i] = imonoKey(k);
  }
}

// ---------------- decode top boxes + sigmoid + validity ----------------------
__global__ void k_decode(const int* __restrict__ topIdx, const float* __restrict__ topLogit,
                         const float* __restrict__ deltas, const float* __restrict__ anchors,
                         float* __restrict__ boxK, float* __restrict__ scoreK,
                         int* __restrict__ validK){
  int t = blockIdx.x*256 + threadIdx.x;
  if (t >= BDIM*PRE_NMS) return;
  int b = t / PRE_NMS;
  int idx = topIdx[t];
  if (idx < 0 || idx >= HWA) idx = 0;
  const float* an = anchors + (size_t)idx*4;
  const float* dl = deltas + (((size_t)b*HWA + idx) << 2);
  float wa = an[2]-an[0], ha = an[3]-an[1];
  float cxa = an[0] + 0.5f*wa, cya = an[1] + 0.5f*ha;
  float dx = dl[0], dy = dl[1];
  float dw = fminf(dl[2], BBOX_CLIP), dh = fminf(dl[3], BBOX_CLIP);
  float cx = dx*wa + cxa, cy = dy*ha + cya;
  float pw = expf(dw)*wa, ph = expf(dh)*ha;
  float x1 = fminf(fmaxf(cx - 0.5f*pw, 0.f), IMG_Wf);
  float y1 = fminf(fmaxf(cy - 0.5f*ph, 0.f), IMG_Hf);
  float x2 = fminf(fmaxf(cx + 0.5f*pw, 0.f), IMG_Wf);
  float y2 = fminf(fmaxf(cy + 0.5f*ph, 0.f), IMG_Hf);
  float sc = 1.f / (1.f + expf(-topLogit[t]));
  int valid = (x2 - x1 >= MIN_SIZE) && (y2 - y1 >= MIN_SIZE) && (sc > 0.f);
  float* ob = boxK + (size_t)t*4;
  ob[0]=x1; ob[1]=y1; ob[2]=x2; ob[3]=y2;
  scoreK[t] = sc;
  validK[t] = valid;
}

// ---------------- greedy NMS + final output, one block per batch -------------
__global__ void k_nms(const float* __restrict__ boxK, const float* __restrict__ scoreK,
                      const int* __restrict__ validK, float* __restrict__ outBoxes,
                      float* __restrict__ outScores){
  const int b = blockIdx.x, tid = threadIdx.x;
  __shared__ float4 bx[PRE_NMS];
  __shared__ int keep[PRE_NMS];
  for (int i=tid; i<PRE_NMS; i+=256){
    const float* p = boxK + ((size_t)b*PRE_NMS + i)*4;
    bx[i] = make_float4(p[0],p[1],p[2],p[3]);
    keep[i] = validK[b*PRE_NMS + i];
  }
  __syncthreads();
  for (int i=0; i<PRE_NMS; ++i){
    if (keep[i]){
      float4 bi = bx[i];
      float areai = (bi.z-bi.x)*(bi.w-bi.y);
      for (int j=i+1+tid; j<PRE_NMS; j+=256){
        if (keep[j]){
          float4 bj = bx[j];
          float ix1 = fmaxf(bi.x,bj.x), iy1 = fmaxf(bi.y,bj.y);
          float ix2 = fminf(bi.z,bj.z), iy2 = fminf(bi.w,bj.w);
          float inter = fmaxf(ix2-ix1,0.f)*fmaxf(iy2-iy1,0.f);
          float areaj = (bj.z-bj.x)*(bj.w-bj.y);
          float iou = inter/(areai+areaj-inter+1e-9f);
          if (iou > NMS_THRESH) keep[j] = 0;
        }
      }
    }
    __syncthreads();
  }
  for (int i=tid; i<PRE_NMS; i+=256){
    float4 v = bx[i]; int kp = keep[i];
    float* ob = outBoxes + ((size_t)b*PRE_NMS + i)*4;
    ob[0] = kp ? v.x : 0.f; ob[1] = kp ? v.y : 0.f;
    ob[2] = kp ? v.z : 0.f; ob[3] = kp ? v.w : 0.f;
    outScores[b*PRE_NMS + i] = kp ? scoreK[b*PRE_NMS + i] : 0.f;
  }
}

extern "C" void kernel_launch(void* const* d_in, const int* in_sizes, int n_in,
                              void* d_out, int out_size, void* d_ws, size_t ws_size,
                              hipStream_t stream){
  (void)in_sizes; (void)n_in; (void)out_size; (void)ws_size;
  const float* features = (const float*)d_in[0];
  const float* conv_w   = (const float*)d_in[1];
  const float* conv_b   = (const float*)d_in[2];
  const float* cls_w    = (const float*)d_in[3];
  const float* cls_b    = (const float*)d_in[4];
  const float* bbox_w   = (const float*)d_in[5];
  const float* bbox_b   = (const float*)d_in[6];
  const float* anchors  = (const float*)d_in[7];

  char* ws = (char*)d_ws;
  size_t o = 0;
  __bf16* featB  = (__bf16*)(ws + o); o += (size_t)BDIM*HW*CDIM*2;
  __bf16* wB     = (__bf16*)(ws + o); o += (size_t)9*CDIM*CDIM*2;
  __bf16* wH     = (__bf16*)(ws + o); o += (size_t)16*CDIM*2;
  float* scoresL = (float*)(ws + o);  o += (size_t)BDIM*HWA*4;
  float* deltasA = (float*)(ws + o);  o += (size_t)BDIM*HWA*16;
  int*   topIdx  = (int*)(ws + o);    o += (size_t)BDIM*PRE_NMS*4;
  float* topLog  = (float*)(ws + o);  o += (size_t)BDIM*PRE_NMS*4;
  float* boxK    = (float*)(ws + o);  o += (size_t)BDIM*PRE_NMS*16;
  float* scoreK  = (float*)(ws + o);  o += (size_t)BDIM*PRE_NMS*4;
  int*   validK  = (int*)(ws + o);    o += (size_t)BDIM*PRE_NMS*4;

  float* outBoxes  = (float*)d_out;
  float* outScores = outBoxes + (size_t)BDIM*PRE_NMS*4;

  dim3 gF(HW/32, CDIM/32, BDIM), bF(32,8);
  k_feat_nhwc<<<gF, bF, 0, stream>>>(features, featB);
  k_prep_wconv<<<(9*CDIM*CDIM + 255)/256, 256, 0, stream>>>(conv_w, wB);
  k_prep_whead<<<(16*CDIM + 255)/256, 256, 0, stream>>>(cls_w, bbox_w, wH);
  k_conv_fused<<<(BDIM*HW)/64, 256, 0, stream>>>(featB, wB, wH, conv_b, cls_b, bbox_b,
                                                 scoresL, deltasA);
  k_topk<<<BDIM, 256, 0, stream>>>(scoresL, topIdx, topLog);
  k_decode<<<(BDIM*PRE_NMS + 255)/256, 256, 0, stream>>>(topIdx, topLog, deltasA, anchors,
                                                         boxK, scoreK, validK);
  k_nms<<<BDIM, 256, 0, stream>>>(boxK, scoreK, validK, outBoxes, outScores);
}
